// TransformerBlock_21354577395901
// MI455X (gfx1250) — matrix operand order repaired and checked by reading, not compiled, not run on hardware
//
#include <hip/hip_runtime.h>
#include <hip/hip_bf16.h>

#define TT 512
#define HID 1024
#define DH 64
#define NH 16
#define NKV 4
#define INTER 1024
#define NE 8
#define QS (NH*DH)      // 1024
#define KVS (NKV*DH)    // 256
#define QKVN (QS+2*KVS) // 1536

typedef __attribute__((ext_vector_type(16))) __bf16 v16bf;
typedef __attribute__((ext_vector_type(8)))  float  v8f;

// ---------- helpers ----------
static __device__ inline unsigned short f2bf(float f) {
    unsigned int u = __float_as_uint(f);
    unsigned int r = u + 0x7FFFu + ((u >> 16) & 1u);   // round-to-nearest-even
    return (unsigned short)(r >> 16);
}

static __device__ inline v8f wmma_bf16(v16bf a, v16bf b, v8f c) {
    return __builtin_amdgcn_wmma_f32_16x16x32_bf16(false, a, false, b, (short)0, c, false, false);
}

// A fragment: 16x32 bf16, M = lane&15, K mapping per ISA table (upper half-wave gets +8 / +24)
static __device__ inline v16bf load_a_frag(const unsigned short (*S)[34], int m_off, int lane) {
    union { unsigned short u[16]; v16bf v; } f;
    int row  = m_off + (lane & 15);
    int kadd = (lane & 16) ? 8 : 0;
#pragma unroll
    for (int j = 0; j < 8; ++j) {
        int kb = ((j & 4) ? 16 : 0) + kadd + 2 * (j & 3);
        f.u[2*j]   = S[row][kb];
        f.u[2*j+1] = S[row][kb+1];
    }
    return f.v;
}

// B fragment: 32x16 bf16, N = lane&15, K mapping per ISA table (upper half-wave gets +8 / +24)
static __device__ inline v16bf load_b_frag(const unsigned short (*S)[34], int n_off, int lane) {
    union { unsigned short u[16]; v16bf v; } f;
    int col  = n_off + (lane & 15);
    int kadd = (lane & 16) ? 8 : 0;
#pragma unroll
    for (int j = 0; j < 8; ++j) {
        int kb = ((j & 4) ? 16 : 0) + kadd + 2 * (j & 3);
        f.u[2*j]   = S[col][kb];
        f.u[2*j+1] = S[col][kb+1];
    }
    return f.v;
}

// Load a 64(row) x 32(k) fp32 tile (row-major, contiguous in k) into LDS as bf16.
static __device__ inline void load_tile_rk_f32(unsigned short (*S)[34], const float* base,
                                               int row0, int k0, int ld, int tid) {
#pragma unroll
    for (int p = 0; p < 2; ++p) {
        int row = p * 32 + (tid >> 3);
        int ks  = (tid & 7) * 4;
        const float4 f = *(const float4*)(base + (size_t)(row0 + row) * ld + k0 + ks);
        S[row][ks]   = f2bf(f.x);
        S[row][ks+1] = f2bf(f.y);
        S[row][ks+2] = f2bf(f.z);
        S[row][ks+3] = f2bf(f.w);
    }
}

// Load a [K,N]-layout weight tile (32 k-rows x 64 n, contiguous in n) into LDS transposed to [n][k].
static __device__ inline void load_tile_kn_f32(unsigned short (*S)[34], const float* base,
                                               int n0, int k0, int ldb, int tid) {
#pragma unroll
    for (int p = 0; p < 2; ++p) {
        int kr = p * 16 + (tid >> 4);
        int ns = (tid & 15) * 4;
        const float4 f = *(const float4*)(base + (size_t)(k0 + kr) * ldb + n0 + ns);
        S[ns][kr]   = f2bf(f.x);
        S[ns+1][kr] = f2bf(f.y);
        S[ns+2][kr] = f2bf(f.z);
        S[ns+3][kr] = f2bf(f.w);
    }
}

// Load 64x32 bf16 tile (already bf16 in memory) into LDS.
static __device__ inline void load_tile_rk_u16(unsigned short (*S)[34], const unsigned short* base,
                                               size_t ld, int tid) {
#pragma unroll
    for (int p = 0; p < 2; ++p) {
        int row = p * 32 + (tid >> 3);
        int ks  = (tid & 7) * 4;
        const ushort4 h = *(const ushort4*)(base + (size_t)row * ld + ks);
        S[row][ks]   = h.x;
        S[row][ks+1] = h.y;
        S[row][ks+2] = h.z;
        S[row][ks+3] = h.w;
    }
}

// ---------- RMSNorm ----------
__global__ void __launch_bounds__(256) k_rmsnorm(const float* __restrict__ x,
                                                 const float* __restrict__ w,
                                                 float* __restrict__ y) {
    int t = blockIdx.x;
    const float* xr = x + (size_t)t * HID;
    float4 v = *(const float4*)(xr + threadIdx.x * 4);
    float ss = v.x*v.x + v.y*v.y + v.z*v.z + v.w*v.w;
#pragma unroll
    for (int off = 16; off; off >>= 1) ss += __shfl_xor(ss, off, 32);
    __shared__ float red[8];
    if ((threadIdx.x & 31) == 0) red[threadIdx.x >> 5] = ss;
    __syncthreads();
    float tot = red[0]+red[1]+red[2]+red[3]+red[4]+red[5]+red[6]+red[7];
    float r = rsqrtf(tot * (1.0f / HID) + 1e-5f);
    float4 wv = *(const float4*)(w + threadIdx.x * 4);
    float4 o; o.x = v.x*r*wv.x; o.y = v.y*r*wv.y; o.z = v.z*r*wv.z; o.w = v.w*r*wv.w;
    *(float4*)(y + (size_t)t * HID + threadIdx.x * 4) = o;
}

// ---------- QKV GEMM: C = A @ W^T + b,  W is [N,K] ----------
__global__ void __launch_bounds__(256) k_gemm_qkv(const float* __restrict__ A,
                                                  const float* __restrict__ W,
                                                  const float* __restrict__ bias,
                                                  float* __restrict__ C) {
    __shared__ unsigned short As[64][34], Bs[64][34];
    const int tid = threadIdx.x, wave = tid >> 5, lane = tid & 31;
    const int m0 = blockIdx.y * 64, n0 = blockIdx.x * 64;
    const int m_off = (wave & 3) * 16, n_off = (wave >> 2) * 32;
    v8f acc0 = {0,0,0,0,0,0,0,0}, acc1 = {0,0,0,0,0,0,0,0};
    for (int k0 = 0; k0 < HID; k0 += 32) {
        load_tile_rk_f32(As, A, m0, k0, HID, tid);
        load_tile_rk_f32(Bs, W, n0, k0, HID, tid);   // [N,K] layout
        __syncthreads();
        v16bf a  = load_a_frag(As, m_off, lane);
        acc0 = wmma_bf16(a, load_b_frag(Bs, n_off,      lane), acc0);
        acc1 = wmma_bf16(a, load_b_frag(Bs, n_off + 16, lane), acc1);
        __syncthreads();
    }
#pragma unroll
    for (int r = 0; r < 8; ++r) {
        int m  = m0 + m_off + r + ((lane & 16) ? 8 : 0);
        int nA = n0 + n_off + (lane & 15);
        C[(size_t)m * QKVN + nA]      = acc0[r] + bias[nA];
        C[(size_t)m * QKVN + nA + 16] = acc1[r] + bias[nA + 16];
    }
}

// ---------- YaRN RoPE, in-place on qkv (q heads 0..15, k heads 16..19) ----------
__global__ void __launch_bounds__(256) k_rope(float* __restrict__ qkv, const int* __restrict__ pos) {
    int idx = blockIdx.x * 256 + threadIdx.x;
    if (idx >= TT * (NH + NKV) * 32) return;
    int d2   = idx & 31;
    int head = (idx >> 5) % (NH + NKV);
    int t    = idx / ((NH + NKV) * 32);
    float p  = (float)pos[t];
    float pf = __powf(150000.0f, (float)(2 * d2) * (1.0f / 64.0f));
    float inv_ex = 1.0f / pf, inv_in = 1.0f / (8.0f * pf);
    float ramp = ((float)d2 - 6.0f) * 0.1f;                 // low=6, high=16
    ramp = fminf(fmaxf(ramp, 0.0f), 1.0f);
    float inv_freq = inv_in * ramp + inv_ex * (1.0f - ramp);
    float ang = p * inv_freq;
    const float mscale = 1.2079441542f;                      // 0.1*ln(8)+1
    float c = __cosf(ang) * mscale, s = __sinf(ang) * mscale;
    float* ptr = qkv + (size_t)t * QKVN + ((head < NH) ? head * DH : QS + (head - NH) * DH);
    float x1 = ptr[d2], x2 = ptr[d2 + 32];
    ptr[d2]      = x1 * c - x2 * s;
    ptr[d2 + 32] = x2 * c + x1 * s;
}

// ---------- sliding-window GQA attention with sink (one wave per (t,h)) ----------
__global__ void __launch_bounds__(256) k_attn(const float* __restrict__ qkv,
                                              const float* __restrict__ sinks,
                                              float* __restrict__ attn) {
    const int wave = threadIdx.x >> 5, lane = threadIdx.x & 31;
    const int pair = blockIdx.x * 8 + wave;    // TT*NH = 8192 pairs
    const int h = pair & (NH - 1);
    const int t = pair >> 4;
    const int kvh = h >> 2;                    // NH/NKV = 4
    const float2 qq = *(const float2*)(qkv + (size_t)t * QKVN + h * DH + 2 * lane);
    const float q0 = qq.x * 0.125f, q1 = qq.y * 0.125f;     // 1/sqrt(64)
    const float* kb = qkv + QS + kvh * DH;
    const float* vb = qkv + QS + KVS + kvh * DH;
    int s_lo = t - 127; if (s_lo < 0) s_lo = 0;
    float m_run = -1e30f, l_run = 0.0f, acc0 = 0.0f, acc1 = 0.0f;
    for (int s = s_lo; s <= t; ++s) {
        const float2 kk = *(const float2*)(kb + (size_t)s * QKVN + 2 * lane);
        float part = q0 * kk.x + q1 * kk.y;
#pragma unroll
        for (int off = 16; off; off >>= 1) part += __shfl_xor(part, off, 32);
        float m_new = fmaxf(m_run, part);
        float corr  = __expf(m_run - m_new);
        float p     = __expf(part - m_new);
        const float2 vv = *(const float2*)(vb + (size_t)s * QKVN + 2 * lane);
        l_run = l_run * corr + p;
        acc0  = acc0 * corr + p * vv.x;
        acc1  = acc1 * corr + p * vv.y;
        m_run = m_new;
    }
    float snk   = sinks[h];
    float m_new = fmaxf(m_run, snk);
    float corr  = __expf(m_run - m_new);
    l_run = l_run * corr + __expf(snk - m_new);
    float inv = 1.0f / l_run;
    acc0 *= corr * inv; acc1 *= corr * inv;
    float2 o; o.x = acc0; o.y = acc1;
    *(float2*)(attn + (size_t)t * QS + h * DH + 2 * lane) = o;
}

// ---------- o_proj GEMM + bias + residual add ----------
__global__ void __launch_bounds__(256) k_gemm_o(const float* __restrict__ A,
                                                const float* __restrict__ W,     // [HID,QS]
                                                const float* __restrict__ bias,
                                                const float* __restrict__ resin,
                                                float* __restrict__ resout) {
    __shared__ unsigned short As[64][34], Bs[64][34];
    const int tid = threadIdx.x, wave = tid >> 5, lane = tid & 31;
    const int m0 = blockIdx.y * 64, n0 = blockIdx.x * 64;
    const int m_off = (wave & 3) * 16, n_off = (wave >> 2) * 32;
    v8f acc0 = {0,0,0,0,0,0,0,0}, acc1 = {0,0,0,0,0,0,0,0};
    for (int k0 = 0; k0 < QS; k0 += 32) {
        load_tile_rk_f32(As, A, m0, k0, QS, tid);
        load_tile_rk_f32(Bs, W, n0, k0, QS, tid);
        __syncthreads();
        v16bf a = load_a_frag(As, m_off, lane);
        acc0 = wmma_bf16(a, load_b_frag(Bs, n_off,      lane), acc0);
        acc1 = wmma_bf16(a, load_b_frag(Bs, n_off + 16, lane), acc1);
        __syncthreads();
    }
#pragma unroll
    for (int r = 0; r < 8; ++r) {
        int m  = m0 + m_off + r + ((lane & 16) ? 8 : 0);
        int nA = n0 + n_off + (lane & 15);
        resout[(size_t)m * HID + nA]      = acc0[r] + bias[nA]      + resin[(size_t)m * HID + nA];
        resout[(size_t)m * HID + nA + 16] = acc1[r] + bias[nA + 16] + resin[(size_t)m * HID + nA + 16];
    }
}

// ---------- router: softmax -> top-2 -> renormalize -> dense comb[T,E] ----------
__global__ void __launch_bounds__(256) k_router(const float* __restrict__ x,
                                                const float* __restrict__ rw,
                                                const float* __restrict__ rb,
                                                float* __restrict__ comb) {
    const int wave = threadIdx.x >> 5, lane = threadIdx.x & 31;
    const int t = blockIdx.x * 8 + wave;
    const float* xr = x + (size_t)t * HID;
    float g[NE];
#pragma unroll
    for (int e = 0; e < NE; ++e) {
        const float* wr = rw + (size_t)e * HID;
        float s = 0.0f;
        for (int d = lane; d < HID; d += 32) s += xr[d] * wr[d];
#pragma unroll
        for (int off = 16; off; off >>= 1) s += __shfl_xor(s, off, 32);
        g[e] = s + rb[e];
    }
    float mx = g[0];
#pragma unroll
    for (int e = 1; e < NE; ++e) mx = fmaxf(mx, g[e]);
    float p[NE];
#pragma unroll
    for (int e = 0; e < NE; ++e) p[e] = __expf(g[e] - mx);
    int i1 = 0; float b1v = p[0];
#pragma unroll
    for (int e = 1; e < NE; ++e) if (p[e] > b1v) { b1v = p[e]; i1 = e; }
    int i2 = -1; float b2v = -1.0f;
#pragma unroll
    for (int e = 0; e < NE; ++e) if (e != i1 && p[e] > b2v) { b2v = p[e]; i2 = e; }
    float inv = 1.0f / (b1v + b2v);
    if (lane < NE)
        comb[(size_t)t * NE + lane] = (lane == i1) ? b1v * inv : ((lane == i2) ? b2v * inv : 0.0f);
}

// ---------- w1 GEMM + bias + SwiGLU + router-weight scale -> act bf16 [E,T,I] ----------
__global__ void __launch_bounds__(256) k_gemm_w1_swiglu(const float* __restrict__ A,
                                                        const float* __restrict__ w1,   // [E,HID,2I]
                                                        const float* __restrict__ b1,   // [E,2I]
                                                        const float* __restrict__ comb,
                                                        unsigned short* __restrict__ act) {
    __shared__ unsigned short As[64][34], Bg[64][34], Bu[64][34];
    const int e = blockIdx.z;
    const float* W  = w1 + (size_t)e * HID * (2 * INTER);
    const float* bb = b1 + (size_t)e * (2 * INTER);
    const int tid = threadIdx.x, wave = tid >> 5, lane = tid & 31;
    const int m0 = blockIdx.y * 64, n0 = blockIdx.x * 64;
    const int m_off = (wave & 3) * 16, n_off = (wave >> 2) * 32;
    v8f g0 = {0,0,0,0,0,0,0,0}, g1 = {0,0,0,0,0,0,0,0};
    v8f u0 = {0,0,0,0,0,0,0,0}, u1 = {0,0,0,0,0,0,0,0};
    for (int k0 = 0; k0 < HID; k0 += 32) {
        load_tile_rk_f32(As, A, m0, k0, HID, tid);
        load_tile_kn_f32(Bg, W, n0,         k0, 2 * INTER, tid);   // gate cols
        load_tile_kn_f32(Bu, W, n0 + INTER, k0, 2 * INTER, tid);   // up cols
        __syncthreads();
        v16bf a = load_a_frag(As, m_off, lane);
        g0 = wmma_bf16(a, load_b_frag(Bg, n_off,      lane), g0);
        g1 = wmma_bf16(a, load_b_frag(Bg, n_off + 16, lane), g1);
        u0 = wmma_bf16(a, load_b_frag(Bu, n_off,      lane), u0);
        u1 = wmma_bf16(a, load_b_frag(Bu, n_off + 16, lane), u1);
        __syncthreads();
    }
#pragma unroll
    for (int r = 0; r < 8; ++r) {
        int m  = m0 + m_off + r + ((lane & 16) ? 8 : 0);
        float cw = comb[(size_t)m * NE + e];
        int nA = n0 + n_off + (lane & 15);
#pragma unroll
        for (int half = 0; half < 2; ++half) {
            int n = nA + half * 16;
            float gv = (half ? g1[r] : g0[r]) + bb[n];
            float uv = (half ? u1[r] : u0[r]) + bb[INTER + n];
            float sl = gv / (1.0f + __expf(-gv));
            act[((size_t)e * TT + m) * INTER + n] = f2bf(cw * sl * uv);
        }
    }
}

// ---------- fused w2 GEMM over K = E*INTER with combined bias -> out ----------
__global__ void __launch_bounds__(256) k_gemm_w2_combine(const unsigned short* __restrict__ act, // [E,T,I] bf16
                                                         const float* __restrict__ w2,  // [E,I,HID] = [8192,1024]
                                                         const float* __restrict__ b2,  // [E,HID]
                                                         const float* __restrict__ comb,
                                                         float* __restrict__ out) {
    __shared__ unsigned short As[64][34], Bs[64][34];
    const int tid = threadIdx.x, wave = tid >> 5, lane = tid & 31;
    const int m0 = blockIdx.y * 64, n0 = blockIdx.x * 64;
    const int m_off = (wave & 3) * 16, n_off = (wave >> 2) * 32;
    v8f acc0 = {0,0,0,0,0,0,0,0}, acc1 = {0,0,0,0,0,0,0,0};
    for (int k0 = 0; k0 < NE * INTER; k0 += 32) {
        const int e = k0 >> 10;
        const int i = k0 & (INTER - 1);
        load_tile_rk_u16(As, act + ((size_t)e * TT + m0) * INTER + i, INTER, tid);
        load_tile_kn_f32(Bs, w2, n0, k0, HID, tid);     // flat [8192,1024] row-major
        __syncthreads();
        v16bf a = load_a_frag(As, m_off, lane);
        acc0 = wmma_bf16(a, load_b_frag(Bs, n_off,      lane), acc0);
        acc1 = wmma_bf16(a, load_b_frag(Bs, n_off + 16, lane), acc1);
        __syncthreads();
    }
#pragma unroll
    for (int r = 0; r < 8; ++r) {
        int m  = m0 + m_off + r + ((lane & 16) ? 8 : 0);
        int nA = n0 + n_off + (lane & 15);
        float bA = 0.0f, bB = 0.0f;
#pragma unroll
        for (int e = 0; e < NE; ++e) {
            float c = comb[(size_t)m * NE + e];
            bA += c * b2[(size_t)e * HID + nA];
            bB += c * b2[(size_t)e * HID + nA + 16];
        }
        out[(size_t)m * HID + nA]      = acc0[r] + bA;
        out[(size_t)m * HID + nA + 16] = acc1[r] + bB;
    }
}

// ---------- host launch ----------
extern "C" void kernel_launch(void* const* d_in, const int* in_sizes, int n_in,
                              void* d_out, int out_size, void* d_ws, size_t ws_size,
                              hipStream_t stream) {
    (void)in_sizes; (void)n_in; (void)out_size; (void)ws_size;
    const float* hidden   = (const float*)d_in[0];
    const int*   positions= (const int*)  d_in[1];
    const float* ln1_w    = (const float*)d_in[2];
    const float* qkv_w    = (const float*)d_in[3];
    const float* qkv_b    = (const float*)d_in[4];
    const float* sinks    = (const float*)d_in[5];
    const float* o_w      = (const float*)d_in[6];
    const float* o_b      = (const float*)d_in[7];
    const float* ln2_w    = (const float*)d_in[8];
    const float* router_w = (const float*)d_in[9];
    const float* router_b = (const float*)d_in[10];
    const float* w1       = (const float*)d_in[11];
    const float* b1       = (const float*)d_in[12];
    const float* w2       = (const float*)d_in[13];
    const float* b2       = (const float*)d_in[14];

    float* out   = (float*)d_out;                 // [T,HID]
    float* resid = out + (size_t)TT * HID;        // [T,HID] (second tuple output)

    char* ws = (char*)d_ws;
    float*          xn1  = (float*)(ws + 0);                                    // 2 MB
    float*          qkv  = (float*)(ws + (2u << 20));                           // 3 MB
    float*          attn = (float*)(ws + (5u << 20));                           // 2 MB
    float*          xn2  = (float*)(ws + (7u << 20));                           // 2 MB
    float*          comb = (float*)(ws + (9u << 20));                           // 16 KB
    unsigned short* act  = (unsigned short*)(ws + (9u << 20) + (1u << 16));     // 8 MB bf16

    // 1. input RMSNorm
    k_rmsnorm<<<TT, 256, 0, stream>>>(hidden, ln1_w, xn1);
    // 2. qkv projection (+bias)
    k_gemm_qkv<<<dim3(QKVN / 64, TT / 64), 256, 0, stream>>>(xn1, qkv_w, qkv_b, qkv);
    // 3. YaRN RoPE in-place on q,k
    k_rope<<<(TT * (NH + NKV) * 32 + 255) / 256, 256, 0, stream>>>(qkv, positions);
    // 4. sliding-window GQA attention with sink
    k_attn<<<TT * NH / 8, 256, 0, stream>>>(qkv, sinks, attn);
    // 5. o_proj (+bias) + residual add -> resid (also second output)
    k_gemm_o<<<dim3(HID / 64, TT / 64), 256, 0, stream>>>(attn, o_w, o_b, hidden, resid);
    // 6. post-attention RMSNorm
    k_rmsnorm<<<TT, 256, 0, stream>>>(resid, ln2_w, xn2);
    // 7. router top-2 weights
    k_router<<<TT / 8, 256, 0, stream>>>(xn2, router_w, router_b, comb);
    // 8. w1 GEMM + SwiGLU + comb scaling -> act (bf16), batched over experts
    k_gemm_w1_swiglu<<<dim3(INTER / 64, TT / 64, NE), 256, 0, stream>>>(xn2, w1, b1, comb, act);
    // 9. fused down-projection over K = E*INTER with combined bias -> out
    k_gemm_w2_combine<<<dim3(HID / 64, TT / 64), 256, 0, stream>>>(act, w2, b2, comb, out);
}
